// LLamaAttention_32573031973249
// MI455X (gfx1250) — matrix operand order repaired and checked by reading, not compiled, not run on hardware
//
#include <hip/hip_runtime.h>

// ---------------------------------------------------------------------------
// LLaMA attention block for gfx1250 (MI455X), wave32 + WMMA f16->f32,
// with CDNA5 async global->LDS copies (ASYNCcnt) and double-buffered GEMM.
// B=2, S=2048, HID=2048, H=16, D=128.
// ---------------------------------------------------------------------------

typedef _Float16 v8h  __attribute__((ext_vector_type(8)));
typedef _Float16 v16h __attribute__((ext_vector_type(16)));
typedef float    v8f  __attribute__((ext_vector_type(8)));

static __device__ __forceinline__ v16h concat8(v8h lo, v8h hi) {
  return __builtin_shufflevector(lo, hi, 0,1,2,3,4,5,6,7,8,9,10,11,12,13,14,15);
}

// CDNA5 async copy: 16B global -> LDS without a VGPR round trip (ASYNCcnt).
static __device__ __forceinline__ void async_ld16(void* lds, const void* g) {
  asm volatile("global_load_async_to_lds_b128 %0, %1, off"
               :: "v"((unsigned)(size_t)lds), "v"(g)
               : "memory");
}
static __device__ __forceinline__ void wait_async0() {
  asm volatile("s_wait_asynccnt 0x0" ::: "memory");
}

constexpr int B_   = 2;
constexpr int S_   = 2048;
constexpr int HID_ = 2048;
constexpr int H_   = 16;
constexpr int D_   = 128;
constexpr int M_   = B_ * S_;   // 4096 rows of activations

// ---------------------------------------------------------------------------
// f32 -> f16 conversion
// ---------------------------------------------------------------------------
__global__ void cvt_f32_f16(const float* __restrict__ src,
                            _Float16* __restrict__ dst, int n) {
  int i = blockIdx.x * blockDim.x + threadIdx.x;
  if (i < n) dst[i] = (_Float16)src[i];
}

// ---------------------------------------------------------------------------
// GEMM: C[M,N] = A[M,K] * Bw[N,K]^T  (both row-major, K contiguous -> NT GEMM)
// Block tile 128x128, BK=32, 256 threads = 8 waves in 2(M) x 4(N),
// each wave computes 64x32 = 4x2 tiles of 16x16 via v_wmma_f32_16x16x32_f16.
// Tiles are staged with global_load_async_to_lds_b128, double-buffered so the
// k+1 transfer overlaps the k-th WMMA chain.
// MODE 0: f16 row-major out.  MODE 1: f16 out in V^T [b,h,d,s] layout.
// MODE 2: f32 row-major out.
// ---------------------------------------------------------------------------
template <int MODE>
__global__ __launch_bounds__(256) void gemm_f16(const _Float16* __restrict__ A,
                                                const _Float16* __restrict__ Bw,
                                                void* __restrict__ outp) {
  constexpr int K      = HID_;
  constexpr int N      = HID_;
  constexpr int LD     = 40;     // 32 + 8 halfs pad (rows stay 16B aligned)
  constexpr int KSTEPS = K / 32;
  __shared__ __attribute__((aligned(16))) _Float16 lA[2][128 * LD];
  __shared__ __attribute__((aligned(16))) _Float16 lB[2][128 * LD];

  const int tid  = threadIdx.x;
  const int lane = tid & 31, l = lane & 15, hs = lane >> 4;
  const int wave = tid >> 5;
  const int wm   = wave >> 2;               // 0..1
  const int wn   = wave & 3;                // 0..3
  const int bm   = blockIdx.y * 128;
  const int bn   = blockIdx.x * 128;

  const int lr = tid >> 1;                  // loader row 0..127
  const int lc = (tid & 1) * 16;            // loader col segment (halfs)
  const int lo = lr * LD + lc;

  const _Float16* aptr = A  + (size_t)(bm + lr) * K + lc;
  const _Float16* bptr = Bw + (size_t)(bn + lr) * K + lc;

  // prologue: async-stage tile 0 into buffer 0
  async_ld16(&lA[0][lo],     aptr);
  async_ld16(&lA[0][lo + 8], aptr + 8);
  async_ld16(&lB[0][lo],     bptr);
  async_ld16(&lB[0][lo + 8], bptr + 8);
  wait_async0();
  __syncthreads();

  v8f acc[4][2] = {};

  for (int kk = 0; kk < KSTEPS; ++kk) {
    const int cur = kk & 1, nxt = cur ^ 1;
    if (kk + 1 < KSTEPS) {                  // overlap: stage tile kk+1
      const _Float16* an = aptr + (size_t)(kk + 1) * 32;
      const _Float16* bn2 = bptr + (size_t)(kk + 1) * 32;
      async_ld16(&lA[nxt][lo],     an);
      async_ld16(&lA[nxt][lo + 8], an + 8);
      async_ld16(&lB[nxt][lo],     bn2);
      async_ld16(&lB[nxt][lo + 8], bn2 + 8);
    }

    v16h af[4];
#pragma unroll
    for (int mt = 0; mt < 4; ++mt) {
      const _Float16* p = &lA[cur][(wm * 64 + mt * 16 + l) * LD];
      af[mt] = concat8(*(const v8h*)(p + hs * 8),
                       *(const v8h*)(p + 16 + hs * 8));
    }
    v16h bf[2];
#pragma unroll
    for (int nt = 0; nt < 2; ++nt) {
      const _Float16* p = &lB[cur][(wn * 32 + nt * 16 + l) * LD];
      bf[nt] = concat8(*(const v8h*)(p + hs * 8), *(const v8h*)(p + 16 + hs * 8));
    }
#pragma unroll
    for (int mt = 0; mt < 4; ++mt)
#pragma unroll
      for (int nt = 0; nt < 2; ++nt)
        acc[mt][nt] = __builtin_amdgcn_wmma_f32_16x16x32_f16(
            false, af[mt], false, bf[nt], (short)0, acc[mt][nt], false, false);

    wait_async0();
    __syncthreads();
  }

#pragma unroll
  for (int mt = 0; mt < 4; ++mt)
#pragma unroll
    for (int nt = 0; nt < 2; ++nt)
#pragma unroll
      for (int r = 0; r < 8; ++r) {
        const int rowg = bm + wm * 64 + mt * 16 + hs * 8 + r;
        const int colg = bn + wn * 32 + nt * 16 + l;
        const float v  = acc[mt][nt][r];
        if (MODE == 0) {
          ((_Float16*)outp)[(size_t)rowg * N + colg] = (_Float16)v;
        } else if (MODE == 1) {
          const int b = rowg >> 11, s = rowg & (S_ - 1);
          const int h = colg >> 7,  d = colg & (D_ - 1);
          ((_Float16*)outp)[(((size_t)(b * H_ + h)) * D_ + d) * S_ + s] = (_Float16)v;
        } else {
          ((float*)outp)[(size_t)rowg * N + colg] = v;
        }
      }
}

// ---------------------------------------------------------------------------
// RoPE applied in place to Q and K (f16), cos/sin tables are f32 [S, D/2].
// ---------------------------------------------------------------------------
__global__ void rope_qk(_Float16* __restrict__ q, _Float16* __restrict__ k,
                        const float* __restrict__ fcos,
                        const float* __restrict__ fsin) {
  const int idx = blockIdx.x * blockDim.x + threadIdx.x;  // < M_*(HID_/2)
  const int row = idx >> 10;            // HID/2 = 1024 pairs per row
  const int p   = idx & 1023;
  const int h   = p >> 6;               // 64 pairs per head
  const int i   = p & 63;
  const int s   = row & (S_ - 1);
  const float c  = fcos[s * 64 + i];
  const float sn = fsin[s * 64 + i];
  const size_t base = (size_t)row * HID_ + h * D_ + 2 * i;

  float qr = (float)q[base], qi = (float)q[base + 1];
  q[base]     = (_Float16)(qr * c - qi * sn);
  q[base + 1] = (_Float16)(qr * sn + qi * c);

  float kr = (float)k[base], ki = (float)k[base + 1];
  k[base]     = (_Float16)(kr * c - ki * sn);
  k[base + 1] = (_Float16)(kr * sn + ki * c);
}

// ---------------------------------------------------------------------------
// Causal flash attention.  Grid: (S/64, B*H).  Block: 128 threads = 4 waves.
// K and V^T tiles staged via async global->LDS; fragment groups preloaded so
// ds loads pipeline ahead of the WMMA chain.
// ---------------------------------------------------------------------------
__global__ __launch_bounds__(128) void flash_attn(const _Float16* __restrict__ q,
                                                  const _Float16* __restrict__ k,
                                                  const _Float16* __restrict__ vT,
                                                  _Float16* __restrict__ o) {
  constexpr int LDK = 136;   // 128 + 8 pad
  constexpr int LDV = 72;    // 64 + 8 pad
  constexpr int LDP = 72;
  __shared__ __attribute__((aligned(16))) _Float16 lK[64 * LDK];
  __shared__ __attribute__((aligned(16))) _Float16 lV[128 * LDV];
  __shared__ __attribute__((aligned(16))) _Float16 lP[4][16 * LDP];

  const int tid  = threadIdx.x;
  const int lane = tid & 31, l = lane & 15, hs = lane >> 4;
  const int wave = tid >> 5;
  const int bh   = blockIdx.y, b = bh >> 4, h = bh & 15;
  const int qb   = blockIdx.x;
  const float scale = 0.08838834764831845f;  // 1/sqrt(128)

  // Q fragments for this wave's 16 rows (contraction dim = head dim, 4x32)
  const int qrow = qb * 64 + wave * 16 + l;
  const _Float16* qp = q + ((size_t)(b * S_ + qrow)) * HID_ + h * D_;
  v16h qf[4];
#pragma unroll
  for (int c = 0; c < 4; ++c)
    qf[c] = concat8(*(const v8h*)(qp + c * 32 + hs * 8),
                    *(const v8h*)(qp + c * 32 + 16 + hs * 8));

  v8f  oa[8] = {};
  float mrow[8], lrow[8];
#pragma unroll
  for (int r = 0; r < 8; ++r) { mrow[r] = -1e30f; lrow[r] = 0.f; }

  const int myrow = qb * 64 + wave * 16 + hs * 8;  // + r

  for (int kb = 0; kb <= qb; ++kb) {
    // --- async-stage K (64x128, row = key) and V^T (128x64, row = d) ---
    {
      const int krow = kb * 64 + (tid >> 1);
      const _Float16* kp = k + ((size_t)(b * S_ + krow)) * HID_ + h * D_ + (tid & 1) * 64;
      _Float16* kd = &lK[(tid >> 1) * LDK + (tid & 1) * 64];
#pragma unroll
      for (int j = 0; j < 8; ++j) async_ld16(kd + j * 8, kp + j * 8);

      const _Float16* vp = vT + ((size_t)bh * D_ + tid) * S_ + kb * 64;
      _Float16* vd = &lV[tid * LDV];
#pragma unroll
      for (int j = 0; j < 8; ++j) async_ld16(vd + j * 8, vp + j * 8);
    }
    wait_async0();
    __syncthreads();

    // --- S = Q * K^T: preload 4 K-fragments per 32-wide d chunk, then WMMA ---
    v8f sa[4] = {};
#pragma unroll
    for (int c = 0; c < 4; ++c) {
      v16h kf[4];
#pragma unroll
      for (int nt = 0; nt < 4; ++nt) {
        const _Float16* kp2 = &lK[(nt * 16 + l) * LDK + c * 32];
        kf[nt] = concat8(*(const v8h*)(kp2 + hs * 8), *(const v8h*)(kp2 + 16 + hs * 8));
      }
#pragma unroll
      for (int nt = 0; nt < 4; ++nt)
        sa[nt] = __builtin_amdgcn_wmma_f32_16x16x32_f16(
            false, qf[c], false, kf[nt], (short)0, sa[nt], false, false);
    }

    // --- causal mask + online softmax ---
#pragma unroll
    for (int r = 0; r < 8; ++r) {
      const int rowq = myrow + r;
      float vals[4], mx = -1e30f;
#pragma unroll
      for (int nt = 0; nt < 4; ++nt) {
        const int colg = kb * 64 + nt * 16 + l;
        float v = sa[nt][r] * scale;
        if (colg > rowq) v += -1e9f;
        vals[nt] = v;
        mx = fmaxf(mx, v);
      }
#pragma unroll
      for (int off = 1; off < 16; off <<= 1)
        mx = fmaxf(mx, __shfl_xor(mx, off, 32));
      const float mN   = fmaxf(mrow[r], mx);
      const float corr = __expf(mrow[r] - mN);
      float sum = 0.f;
#pragma unroll
      for (int nt = 0; nt < 4; ++nt) {
        const float pv = __expf(vals[nt] - mN);
        sum += pv;
        lP[wave][(hs * 8 + r) * LDP + nt * 16 + l] = (_Float16)pv;
      }
#pragma unroll
      for (int off = 1; off < 16; off <<= 1)
        sum += __shfl_xor(sum, off, 32);
      lrow[r] = lrow[r] * corr + sum;
      mrow[r] = mN;
#pragma unroll
      for (int dt = 0; dt < 8; ++dt) oa[dt][r] *= corr;
    }

    // --- O += P * V: preload V-fragments in groups of 4 ---
#pragma unroll
    for (int c2 = 0; c2 < 2; ++c2) {
      const _Float16* pp = &lP[wave][l * LDP + c2 * 32];
      v16h pf = concat8(*(const v8h*)(pp + hs * 8),
                        *(const v8h*)(pp + 16 + hs * 8));
#pragma unroll
      for (int g = 0; g < 2; ++g) {
        v16h vf[4];
#pragma unroll
        for (int j = 0; j < 4; ++j) {
          const _Float16* vp2 = &lV[((g * 4 + j) * 16 + l) * LDV + c2 * 32];
          vf[j] = concat8(*(const v8h*)(vp2 + hs * 8), *(const v8h*)(vp2 + 16 + hs * 8));
        }
#pragma unroll
        for (int j = 0; j < 4; ++j)
          oa[g * 4 + j] = __builtin_amdgcn_wmma_f32_16x16x32_f16(
              false, pf, false, vf[j], (short)0, oa[g * 4 + j], false, false);
      }
    }
    __syncthreads();
  }

  // --- normalize and write out (f16, [b,s,h*D+d] layout) ---
#pragma unroll
  for (int dt = 0; dt < 8; ++dt)
#pragma unroll
    for (int r = 0; r < 8; ++r) {
      const int row = myrow + r;
      const int col = h * D_ + dt * 16 + l;
      o[((size_t)(b * S_ + row)) * HID_ + col] = (_Float16)(oa[dt][r] / lrow[r]);
    }
}

// ---------------------------------------------------------------------------
// Host side
// ---------------------------------------------------------------------------
extern "C" void kernel_launch(void* const* d_in, const int* in_sizes, int n_in,
                              void* d_out, int out_size, void* d_ws, size_t ws_size,
                              hipStream_t stream) {
  (void)in_sizes; (void)n_in; (void)out_size; (void)ws_size;

  const float* x  = (const float*)d_in[0];
  const float* wq = (const float*)d_in[1];
  const float* wk = (const float*)d_in[2];
  const float* wv = (const float*)d_in[3];
  const float* wo = (const float*)d_in[4];
  const float* fc = (const float*)d_in[5];
  const float* fs = (const float*)d_in[6];
  // d_in[7] = mask (causal mask reproduced analytically), d_in[8] = pos_start (0)

  const size_t NX = (size_t)M_ * HID_;     // 8388608 halfs
  const size_t NW = (size_t)HID_ * HID_;   // 4194304 halfs

  _Float16* w   = (_Float16*)d_ws;
  _Float16* xh  = w;  w += NX;
  _Float16* wqh = w;  w += NW;
  _Float16* wkh = w;  w += NW;
  _Float16* wvh = w;  w += NW;
  _Float16* woh = w;  w += NW;
  _Float16* qh  = w;  w += NX;
  _Float16* kh  = w;  w += NX;
  _Float16* vt  = w;  w += NX;   // V^T per head: [b,h,d,s]
  _Float16* oh  = w;  w += NX;

  cvt_f32_f16<<<(int)(NX / 256), 256, 0, stream>>>(x,  xh,  (int)NX);
  cvt_f32_f16<<<(int)(NW / 256), 256, 0, stream>>>(wq, wqh, (int)NW);
  cvt_f32_f16<<<(int)(NW / 256), 256, 0, stream>>>(wk, wkh, (int)NW);
  cvt_f32_f16<<<(int)(NW / 256), 256, 0, stream>>>(wv, wvh, (int)NW);
  cvt_f32_f16<<<(int)(NW / 256), 256, 0, stream>>>(wo, woh, (int)NW);

  dim3 gg(HID_ / 128, M_ / 128);  // (16, 32)
  gemm_f16<0><<<gg, 256, 0, stream>>>(xh, wqh, qh);
  gemm_f16<0><<<gg, 256, 0, stream>>>(xh, wkh, kh);
  gemm_f16<1><<<gg, 256, 0, stream>>>(xh, wvh, vt);

  rope_qk<<<(M_ * (HID_ / 2)) / 256, 256, 0, stream>>>(qh, kh, fc, fs);

  flash_attn<<<dim3(S_ / 64, B_ * H_), 128, 0, stream>>>(qh, kh, vt, oh);

  gemm_f16<2><<<gg, 256, 0, stream>>>(oh, woh, d_out);
}
